// Seq2Seq_38972533244558
// MI455X (gfx1250) — compile-verified, matrix operand order checked by reading, not run on hardware
//
#include <hip/hip_runtime.h>

// ---------------------------------------------------------------------------
// GRU seq2seq for MI455X (gfx1250).
//  - bf16 WMMA (v_wmma_f32_16x16x32_bf16) for all GEMMs, f32 accumulate.
//  - 64 blocks x 256 threads (8 waves, wave32): M=32 batch tile per block,
//    each wave owns 128 gate columns and BOTH 16-row M tiles so every
//    B-fragment read from L2 feeds 2 WMMAs (halves L2 weight traffic vs M=16).
//  - Hidden state double-buffered as bf16 inside the LDS A panel
//    [x(160) | h_even(1024) | h_odd(1024)], plus an f32 master copy with
//    per-lane ownership (no races, one barrier per step).
// ---------------------------------------------------------------------------

typedef __attribute__((ext_vector_type(16))) __bf16 v16bf;
typedef __attribute__((ext_vector_type(8)))  float  v8f;

#define B_TILE 32
#define POSE_D 135
#define RNN_D  1024
#define NGATE  3072            // 3 * RNN
#define KX     160             // x-region of concatenated K (135 padded to 160)
#define KTOT   1184            // 160 + 1024 (packed weight row length)
#define A_S    2208            // LDS A row stride: 160 + 1024 + 1024 (32B-aligned rows)
#define SEQ_T  144             // SEED_LEN + TGT_LEN
#define T_ENC  119             // SEED_LEN - 1 encoder steps
#define T_DEC  24
#define NBLK   64              // 2048 / B_TILE
#define NFC    144             // POSE padded to 9 N-tiles
#define NTHR   256

#define WMMA_BF16(a, b, c) \
  __builtin_amdgcn_wmma_f32_16x16x32_bf16(false, (a), false, (b), (short)0, (c), false, false)

__device__ __forceinline__ float sigm(float x) { return 1.0f / (1.0f + __expf(-x)); }

// Pack [Wih | zeros | Whh] -> bf16, row n (gate col), k in [0,KTOT)
__global__ void build_wcat(const float* __restrict__ Wih, const float* __restrict__ Whh,
                           __bf16* __restrict__ wcat) {
  int i = blockIdx.x * blockDim.x + threadIdx.x;
  if (i >= NGATE * KTOT) return;
  int n = i / KTOT, k = i % KTOT;
  float v = 0.0f;
  if (k < POSE_D)   v = Wih[n * POSE_D + k];
  else if (k >= KX) v = Whh[n * RNN_D + (k - KX)];
  wcat[i] = (__bf16)v;
}

// Pack Wfc (135x1024) -> bf16 144x1024 (N padded with zero rows)
__global__ void build_wfc(const float* __restrict__ Wfc, __bf16* __restrict__ wfcb) {
  int i = blockIdx.x * blockDim.x + threadIdx.x;
  if (i >= NFC * RNN_D) return;
  int n = i / RNN_D, k = i % RNN_D;
  wfcb[i] = (__bf16)(n < POSE_D ? Wfc[n * RNN_D + k] : 0.0f);
}

// bias[0..1023]=b_r, [1024..]=b_z, [2048..]=b_in, [3072..]=b_hn
__global__ void build_bias(const float* __restrict__ bih, const float* __restrict__ bhh,
                           float* __restrict__ bias) {
  int c = blockIdx.x * blockDim.x + threadIdx.x;
  if (c >= RNN_D) return;
  bias[c]             = bih[c] + bhh[c];
  bias[RNN_D + c]     = bih[RNN_D + c] + bhh[RNN_D + c];
  bias[2 * RNN_D + c] = bih[2 * RNN_D + c];
  bias[3 * RNN_D + c] = bhh[2 * RNN_D + c];
}

__global__ __launch_bounds__(NTHR, 1) void gru_seq2seq(
    const float* __restrict__ poses, const __bf16* __restrict__ wcat,
    const __bf16* __restrict__ wfcb, const float* __restrict__ bias,
    const float* __restrict__ bfc, float* __restrict__ out) {
  extern __shared__ char smem[];
  __bf16* A  = (__bf16*)smem;                          // [32][A_S]: [x | h_even | h_odd]
  float*  H  = (float*)(smem + B_TILE * A_S * 2);      // [32][1024] f32 hidden master
  float*  IO = H + B_TILE * RNN_D;                     // [32][144]  decoder inp/out

  const int tid   = threadIdx.x;
  const int wave  = tid >> 5;
  const int lane  = tid & 31;
  const int b0    = blockIdx.x * B_TILE;
  const int mrow  = (lane >> 4) * 8;       // C/D tile: lane half -> rows 0-7 / 8-15
  const int ncol  = lane & 15;             // column within a 16-wide N tile
  const int khalf = (lane >> 4) * 16;      // A/B frag: lane half covers 16 of 32 K

  for (int i = tid; i < B_TILE * A_S;   i += NTHR) A[i] = (__bf16)0.0f;
  for (int i = tid; i < B_TILE * RNN_D; i += NTHR) H[i] = 0.0f;
  __syncthreads();

  for (int t = 0; t < T_ENC + T_DEC; ++t) {
    // ---- stage step input x into A[:, 0:135] as bf16 -----------------------
    if (t <= T_ENC) {
      for (int i = tid; i < B_TILE * POSE_D; i += NTHR) {
        int m = i / POSE_D, k = i % POSE_D;
        float xv = poses[((size_t)(b0 + m) * SEQ_T + t) * POSE_D + k];
        A[m * A_S + k] = (__bf16)xv;
        if (t == T_ENC) IO[m * NFC + k] = xv;   // decoder inp0 = poses[:,119,:]
      }
    } else {
      for (int i = tid; i < B_TILE * POSE_D; i += NTHR) {
        int m = i / POSE_D, k = i % POSE_D;
        A[m * A_S + k] = (__bf16)IO[m * NFC + k];
      }
    }
    __syncthreads();

    const int hcur = KX + (t & 1) * RNN_D;         // bf16 h read region
    const int hnxt = KX + ((t + 1) & 1) * RNN_D;   // bf16 h write region

    // ---- gate GEMMs: wave owns 128 gate cols x both M tiles ----------------
    for (int cp = 0; cp < 8; ++cp) {               // 8 N-tiles of 16 cols each
      const int n = wave * 128 + cp * 16 + ncol;   // this lane's gate column
      const __bf16* Br  = wcat + (size_t)n * KTOT + khalf;
      const __bf16* Bz  = wcat + (size_t)(RNN_D + n) * KTOT + khalf;
      const __bf16* Bn  = wcat + (size_t)(2 * RNN_D + n) * KTOT + khalf;
      const __bf16* A0x = A + (size_t)ncol * A_S + khalf;          // M rows 0-15, x region
      const __bf16* A1x = A0x + 16 * A_S;                          // M rows 16-31
      const __bf16* A0h = A + (size_t)ncol * A_S + hcur + khalf;   // h region
      const __bf16* A1h = A0h + 16 * A_S;

      v8f ar0 = {}, az0 = {}, ain0 = {}, ahn0 = {};
      v8f ar1 = {}, az1 = {}, ain1 = {}, ahn1 = {};

      // x-part (K = 0..159): feeds r, z and the i_n accumulator
#pragma unroll 1
      for (int k0 = 0; k0 < KX; k0 += 32) {
        v16bf br = *(const v16bf*)(Br + k0);
        v16bf bz = *(const v16bf*)(Bz + k0);
        v16bf bn = *(const v16bf*)(Bn + k0);
        v16bf a0 = *(const v16bf*)(A0x + k0);
        v16bf a1 = *(const v16bf*)(A1x + k0);
        ar0 = WMMA_BF16(a0, br, ar0);  ar1 = WMMA_BF16(a1, br, ar1);
        az0 = WMMA_BF16(a0, bz, az0);  az1 = WMMA_BF16(a1, bz, az1);
        ain0 = WMMA_BF16(a0, bn, ain0); ain1 = WMMA_BF16(a1, bn, ain1);
      }
      // h-part (K = 160..1183): feeds r, z and the h_n accumulator
#pragma unroll 1
      for (int k0 = 0; k0 < RNN_D; k0 += 32) {
        v16bf br = *(const v16bf*)(Br + KX + k0);
        v16bf bz = *(const v16bf*)(Bz + KX + k0);
        v16bf bn = *(const v16bf*)(Bn + KX + k0);
        v16bf a0 = *(const v16bf*)(A0h + k0);
        v16bf a1 = *(const v16bf*)(A1h + k0);
        ar0 = WMMA_BF16(a0, br, ar0);  ar1 = WMMA_BF16(a1, br, ar1);
        az0 = WMMA_BF16(a0, bz, az0);  az1 = WMMA_BF16(a1, bz, az1);
        ahn0 = WMMA_BF16(a0, bn, ahn0); ahn1 = WMMA_BF16(a1, bn, ahn1);
      }

      const float brv = bias[n],            bzv  = bias[RNN_D + n];
      const float binv = bias[2 * RNN_D + n], bhnv = bias[3 * RNN_D + n];
      // GRU update; (m,n) strictly lane-owned: H read-then-write is race-free,
      // A h_next writes are disjoint from everyone's h_cur reads.
#pragma unroll
      for (int v = 0; v < 8; ++v) {
        int m = mrow + v;
        float r  = sigm(ar0[v] + brv);
        float z  = sigm(az0[v] + bzv);
        float nn = tanhf(ain0[v] + binv + r * (ahn0[v] + bhnv));
        float hv = (1.0f - z) * nn + z * H[m * RNN_D + n];
        H[m * RNN_D + n] = hv;
        A[m * A_S + hnxt + n] = (__bf16)hv;
      }
#pragma unroll
      for (int v = 0; v < 8; ++v) {
        int m = 16 + mrow + v;
        float r  = sigm(ar1[v] + brv);
        float z  = sigm(az1[v] + bzv);
        float nn = tanhf(ain1[v] + binv + r * (ahn1[v] + bhnv));
        float hv = (1.0f - z) * nn + z * H[m * RNN_D + n];
        H[m * RNN_D + n] = hv;
        A[m * A_S + hnxt + n] = (__bf16)hv;
      }
    }
    __syncthreads();   // h_next fully committed

    // ---- decoder head: out = inp + h @ Wfc^T + bfc -------------------------
    if (t >= T_ENC) {
      for (int j = wave; j < 18; j += 8) {        // 2 M-tiles x 9 N-tiles
        const int mt = j / 9, nt = j % 9;
        const int n  = nt * 16 + ncol;
        const __bf16* Bf = wfcb + (size_t)n * RNN_D + khalf;
        const __bf16* Ah = A + (size_t)(mt * 16 + ncol) * A_S + hnxt + khalf;
        v8f acc = {};
#pragma unroll 2
        for (int k0 = 0; k0 < RNN_D; k0 += 32) {
          v16bf af = *(const v16bf*)(Ah + k0);
          v16bf bf = *(const v16bf*)(Bf + k0);
          acc = WMMA_BF16(af, bf, acc);
        }
        if (n < POSE_D) {
          const float bv = bfc[n];
          const int td = t - T_ENC;
#pragma unroll
          for (int v = 0; v < 8; ++v) {
            const int m = mt * 16 + mrow + v;
            float o = IO[m * NFC + n] + acc[v] + bv;
            IO[m * NFC + n] = o;    // becomes next step's inp
            out[((size_t)(b0 + m) * T_DEC + td) * POSE_D + n] = o;
          }
        }
      }
      __syncthreads();
    }
  }
}

extern "C" void kernel_launch(void* const* d_in, const int* in_sizes, int n_in,
                              void* d_out, int out_size, void* d_ws, size_t ws_size,
                              hipStream_t stream) {
  (void)in_sizes; (void)n_in; (void)out_size; (void)ws_size;
  const float* poses = (const float*)d_in[0];
  const float* Wih   = (const float*)d_in[1];
  const float* Whh   = (const float*)d_in[2];
  const float* bih   = (const float*)d_in[3];
  const float* bhh   = (const float*)d_in[4];
  const float* Wfc   = (const float*)d_in[5];
  const float* bfc   = (const float*)d_in[6];
  float* out = (float*)d_out;

  // workspace: wcat (7.27 MB bf16) | wfcb (288 KB bf16) | bias (16 KB f32)
  char* ws = (char*)d_ws;
  __bf16* wcat = (__bf16*)ws;
  __bf16* wfcb = (__bf16*)(ws + (size_t)NGATE * KTOT * 2);
  float*  bias = (float*)(ws + (size_t)NGATE * KTOT * 2 + (size_t)NFC * RNN_D * 2);

  build_wcat<<<(NGATE * KTOT + 255) / 256, 256, 0, stream>>>(Wih, Whh, wcat);
  build_wfc<<<(NFC * RNN_D + 255) / 256, 256, 0, stream>>>(Wfc, wfcb);
  build_bias<<<(RNN_D + 255) / 256, 256, 0, stream>>>(bih, bhh, bias);

  size_t smem = (size_t)B_TILE * A_S * 2      // A panel (bf16, x + 2x h buffers)
              + (size_t)B_TILE * RNN_D * 4    // H master (f32)
              + (size_t)B_TILE * NFC * 4;     // IO (f32)  => ~284 KB < 320 KB WGP LDS
  gru_seq2seq<<<NBLK, NTHR, smem, stream>>>(poses, wcat, wfcb, bias, bfc, out);
}
